// ScoreNetLSTM_17729624998472
// MI455X (gfx1250) — compile-verified, matrix operand order checked by reading, not run on hardware
//
#include <hip/hip_runtime.h>

// Problem constants
#define BB 512   // batch
#define TT 512   // timesteps
#define DD 256   // input dim
#define HH 200   // hidden
#define G4 800   // 4*H

typedef __attribute__((ext_vector_type(16))) __bf16   v16bf;
typedef __attribute__((ext_vector_type(8)))  float    v8f;
typedef __attribute__((ext_vector_type(4)))  unsigned u32x4;
typedef int int4v __attribute__((vector_size(16)));   // async-to-LDS operand

// TDM descriptor group types (clang-23 6-arg builtin form)
typedef __attribute__((ext_vector_type(4))) unsigned tdm_g0_t;
typedef __attribute__((ext_vector_type(8))) int      tdm_g1_t;
typedef __attribute__((ext_vector_type(4))) int      tdm_g2_t;

#define AS1 __attribute__((address_space(1)))
#define AS3 __attribute__((address_space(3)))

#if __has_builtin(__builtin_amdgcn_global_load_async_to_lds_b128)
#define HAVE_ASYNC_LDS 1
#endif
#if __has_builtin(__builtin_amdgcn_tensor_load_to_lds)
#define HAVE_TDM 1
#endif

union Frag16 { u32x4 q[2]; v16bf v; };

__device__ __forceinline__ v8f wmma_bf16(v16bf a, v16bf b, v8f c) {
  // D(f32 16x16) = A(bf16 16x32) * B(bf16 32x16) + C
  return __builtin_amdgcn_wmma_f32_16x16x32_bf16(
      /*neg_a=*/false, a, /*neg_b=*/false, b,
      /*c_mod=*/(short)0, c, /*reuse_a=*/false, /*reuse_b=*/false);
}

// 16-lane-striped bf16 fragment load (A: lane&15 = M row; B: lane&15 = N col
// of row-major [N][K] weights).  lanes 0-15: K=kb..kb+7 and kb+16..kb+23 with
// kb = ks*32; lanes 16-31: same with kb += 8.
__device__ __forceinline__ v16bf load_frag(const __bf16* p) {
  Frag16 f;
  f.q[0] = *(const u32x4*)(p);
  f.q[1] = *(const u32x4*)(p + 16);
  return f.v;
}

__device__ __forceinline__ void async_copy16(const void* g, void* l) {
#ifdef HAVE_ASYNC_LDS
  __builtin_amdgcn_global_load_async_to_lds_b128(
      (AS1 int4v*)(uintptr_t)g,
      (AS3 int4v*)(unsigned)(uintptr_t)l, 0, 0);
#else
  (void)g; (void)l;
#endif
}

__device__ __forceinline__ void wait_async0() {
#if __has_builtin(__builtin_amdgcn_s_wait_asynccnt)
  __builtin_amdgcn_s_wait_asynccnt(0);
#else
  asm volatile("s_wait_asynccnt 0x0" ::: "memory");
#endif
}

__device__ __forceinline__ void wait_tensor0() {
#if __has_builtin(__builtin_amdgcn_s_wait_tensorcnt)
  __builtin_amdgcn_s_wait_tensorcnt(0);
#else
  asm volatile("s_wait_tensorcnt 0x0" ::: "memory");
#endif
}

__device__ __forceinline__ float sigmoid_f(float x) {
  return 1.0f / (1.0f + __expf(-x));
}
__device__ __forceinline__ float tanh_f(float x) {
  float ax = fabsf(x);
  float e  = __expf(-2.0f * ax);
  float t  = (1.0f - e) / (1.0f + e);
  return copysignf(t, x);
}

// ---------------------------------------------------------------- prep kernels
__global__ void f32_to_bf16_kernel(const float* __restrict__ s,
                                   __bf16* __restrict__ d, int n) {
  int i = blockIdx.x * blockDim.x + threadIdx.x;
  int stride = gridDim.x * blockDim.x;
  for (; i < n; i += stride) d[i] = (__bf16)s[i];
}

__global__ void bias_sum_kernel(const float* __restrict__ a,
                                const float* __restrict__ b,
                                float* __restrict__ o, int n) {
  int i = blockIdx.x * blockDim.x + threadIdx.x;
  if (i < n) o[i] = a[i] + b[i];
}

// Store one 16x16 f32 C tile to pre[T][B][800]: lane=column n, vgpr r -> m.
__device__ __forceinline__ void store_tile(float* __restrict__ pre,
                                           size_t row0, int nt, int col,
                                           int hi, v8f acc) {
#pragma unroll
  for (int r = 0; r < 8; ++r) {
    size_t g = row0 + r + 8 * hi;          // global x row = b*T + t
    size_t bidx = g >> 9;                  // / T
    size_t t    = g & 511;                 // % T
    pre[(t * BB + bidx) * G4 + nt * 16 + col] = acc[r];
  }
}

// ------------------------------------------------- kernel 1: pre = x@W_ih^T+b
// xb: [B*T][256] bf16 (row = b*T + t), Wihb: [800][256] bf16,
// pre: [T][B][800] f32 (recurrence-friendly layout).
// Each wave owns 5 of the 50 N tiles; tiles are processed in PAIRS sharing the
// A fragment so two independent WMMA chains can hide B-load latency.
__global__ __launch_bounds__(320) void pre_gemm_kernel(
    const __bf16* __restrict__ xb,
    const __bf16* __restrict__ Wihb,
    const float*  __restrict__ biasSum,
    float*        __restrict__ pre) {
  __shared__ __attribute__((aligned(16))) __bf16 alds[16 * 264];  // pad 264

  const int tid  = threadIdx.x;
  const int wave = tid >> 5;
  const int lane = tid & 31;
  const int mt   = blockIdx.x;          // M tile (16 rows of x)
  const size_t row0 = (size_t)mt * 16;

  // Stage A tile 16x256 bf16 into LDS (512 chunks of 16B) via async-to-LDS
#ifdef HAVE_ASYNC_LDS
  for (int i = tid; i < 512; i += 320) {
    int r = i >> 5, ck = i & 31;
    async_copy16(&xb[(row0 + r) * DD + ck * 8], &alds[r * 264 + ck * 8]);
  }
  wait_async0();
#else
  for (int i = tid; i < 512; i += 320) {
    int r = i >> 5, ck = i & 31;
    *(u32x4*)&alds[r * 264 + ck * 8] =
        *(const u32x4*)&xb[(row0 + r) * DD + ck * 8];
  }
#endif
  __syncthreads();

  const int col = lane & 15;       // N column for B/C frags, M row for A frag
  const int hi  = lane >> 4;       // half-wave select
  const int kb8 = hi << 3;

  // ---- two paired iterations: tiles (w, w+10) and (w+20, w+30) ----
#pragma unroll
  for (int pair = 0; pair < 2; ++pair) {
    const int nt0 = wave + pair * 20;
    const int nt1 = nt0 + 10;
    float bias0 = biasSum[nt0 * 16 + col];
    float bias1 = biasSum[nt1 * 16 + col];
    v8f acc0 = {bias0, bias0, bias0, bias0, bias0, bias0, bias0, bias0};
    v8f acc1 = {bias1, bias1, bias1, bias1, bias1, bias1, bias1, bias1};
    const __bf16* w0 = &Wihb[(size_t)(nt0 * 16 + col) * DD + kb8];
    const __bf16* w1 = &Wihb[(size_t)(nt1 * 16 + col) * DD + kb8];
#pragma unroll
    for (int ks = 0; ks < 8; ++ks) {       // K = 256 = 8 * 32
      v16bf a  = load_frag(&alds[col * 264 + ks * 32 + kb8]);
      v16bf b0 = load_frag(w0 + ks * 32);
      v16bf b1 = load_frag(w1 + ks * 32);
      acc0 = wmma_bf16(a, b0, acc0);       // two independent chains
      acc1 = wmma_bf16(a, b1, acc1);
    }
    store_tile(pre, row0, nt0, col, hi, acc0);
    store_tile(pre, row0, nt1, col, hi, acc1);
  }
  // ---- tail tile: nt = wave + 40 ----
  {
    const int nt = wave + 40;
    float bias = biasSum[nt * 16 + col];
    v8f acc = {bias, bias, bias, bias, bias, bias, bias, bias};
    const __bf16* w = &Wihb[(size_t)(nt * 16 + col) * DD + kb8];
#pragma unroll
    for (int ks = 0; ks < 8; ++ks) {
      v16bf a = load_frag(&alds[col * 264 + ks * 32 + kb8]);
      v16bf b = load_frag(w + ks * 32);
      acc = wmma_bf16(a, b, acc);
    }
    store_tile(pre, row0, nt, col, hi, acc);
  }
}

// --------------------------------------------- kernel 2: recurrence + output
// One workgroup per 16 batch rows.  W_hh lives in LDS for the whole kernel
// (staged once by the Tensor Data Mover).  13 waves; wave jt owns hidden
// columns j = jt*16 .. jt*16+15 and computes the i/f/g/o gate tiles for those
// columns => the c-state stays in registers for all 512 steps.
__global__ __launch_bounds__(416) void lstm_kernel(
    const __bf16* __restrict__ Whhb,   // [800][200] bf16 row-major
    const float*  __restrict__ pre,    // [T][B][800] f32
    const float*  __restrict__ Wout,   // [T*H] f32
    const float*  __restrict__ bout,   // [1]
    float*        __restrict__ out) {  // [B]
  __shared__ __attribute__((aligned(16))) __bf16 wlds[G4 * HH];   // 320000 B
  __shared__ __attribute__((aligned(16))) __bf16 hlds[16 * 232];  //   7424 B
  __shared__ float oacc[16];                                      //     64 B

  const int tid  = threadIdx.x;
  const int wave = tid >> 5;          // 0..12 == j-tile
  const int lane = tid & 31;
  const int b0   = blockIdx.x * 16;

  // ---- Stage W_hh (800x200 bf16, contiguous) into LDS ----
#ifdef HAVE_TDM
  if (wave == 0) {
    // Tensor DMA descriptor (D#), 2D tensor == tile, data_size = 2 bytes.
    unsigned lds_base = (unsigned)(uintptr_t)&wlds[0];
    unsigned long long ga = (unsigned long long)(uintptr_t)Whhb;
    tdm_g0_t g0;
    g0[0] = 1u;                                   // count=1, user descriptor
    g0[1] = lds_base;                             // lds_addr[31:0]
    g0[2] = (unsigned)(ga & 0xFFFFFFFFu);         // global_addr[31:0]
    g0[3] = (unsigned)((ga >> 32) & 0x01FFFFFFu)  // global_addr[56:32]
            | (2u << 30);                         // type=2 ("image")
    tdm_g1_t g1;
    g1[0] = (int)(1u << 16);        // workgroup_mask=0, data_size=1 (2B)
    g1[1] = (int)((unsigned)HH << 16);   // tensor_dim0[15:0]=200 @bits[79:48]
    g1[2] = (int)((unsigned)G4 << 16);   // tensor_dim1[15:0]=800 @bits[111:80]
    g1[3] = (int)((unsigned)HH << 16);   // tile_dim0=200 @bits[127:112]
    g1[4] = (int)G4;                     // tile_dim1=800 @bits[143:128]
    g1[5] = (int)HH;                     // tensor_dim0_stride=200 @bits[207:160]
    g1[6] = (int)(((unsigned)(HH * G4) & 0xFFFFu) << 16);  // dim1_stride lo16
    g1[7] = (int)((unsigned)(HH * G4) >> 16);              // dim1_stride hi
    tdm_g2_t gz;
    gz[0] = 0; gz[1] = 0; gz[2] = 0; gz[3] = 0;   // groups 2/3 unused (2D)
    tdm_g1_t ge;                                   // extra group (clang-23 form)
    ge[0] = 0; ge[1] = 0; ge[2] = 0; ge[3] = 0;
    ge[4] = 0; ge[5] = 0; ge[6] = 0; ge[7] = 0;
    __builtin_amdgcn_tensor_load_to_lds(g0, g1, gz, gz, ge, 0);
    wait_tensor0();
  }
#elif defined(HAVE_ASYNC_LDS)
  for (int i = tid; i < (G4 * HH * 2) / 16; i += 416)
    async_copy16((const char*)Whhb + i * 16, (char*)wlds + i * 16);
  wait_async0();
#else
  {
    const u32x4* src = (const u32x4*)Whhb;
    u32x4*       dst = (u32x4*)wlds;
    for (int i = tid; i < (G4 * HH * 2) / 16; i += 416) dst[i] = src[i];
  }
#endif
  // h = 0 (including K-pad columns 200..231)
  for (int i = tid; i < (16 * 232) / 2; i += 416) ((unsigned*)hlds)[i] = 0u;
  if (tid < 16) oacc[tid] = 0.0f;
  __syncthreads();

  const int jt  = wave;
  const int col = lane & 15;
  const int hi  = lane >> 4;
  const int kb8 = hi << 3;
  const int j   = jt * 16 + col;
  const bool jok = (j < HH);

  float c[8], op[8];
#pragma unroll
  for (int r = 0; r < 8; ++r) { c[r] = 0.0f; op[r] = 0.0f; }

  // Double-buffered pre fragments (C-layout match: lane=col n, vgpr r -> m)
  float pr[4][8], prn[4][8];
#pragma unroll
  for (int g = 0; g < 4; ++g)
#pragma unroll
    for (int r = 0; r < 8; ++r)
      pr[g][r] = jok ? pre[((size_t)b0 + r + 8 * hi) * G4 + g * HH + j] : 0.0f;

  const __bf16* wbase = jok ? &wlds[(size_t)j * HH] : &wlds[0];

  for (int t = 0; t < TT; ++t) {
    // ---- GEMM: acc[g] = h @ W_hh[g-block]^T (reads hlds of step t) ----
    v8f acc[4];
#pragma unroll
    for (int g = 0; g < 4; ++g) acc[g] = (v8f){0, 0, 0, 0, 0, 0, 0, 0};

#pragma unroll
    for (int ks = 0; ks < 7; ++ks) {         // K = 200 padded to 224
      v16bf a = load_frag(&hlds[col * 232 + ks * 32 + kb8]);
      // Load the 4 gate B-fragments up front, then issue the 4 WMMAs.
      Frag16 f[4];
      if (ks < 6) {                           // compile-time (unrolled)
#pragma unroll
        for (int g = 0; g < 4; ++g) {
          const __bf16* p = wbase + (size_t)g * HH * HH + ks * 32 + kb8;
          f[g].q[0] = *(const u32x4*)(p);
          f[g].q[1] = *(const u32x4*)(p + 16);
        }
        if (!jok) {
#pragma unroll
          for (int g = 0; g < 4; ++g) {
            f[g].q[0] = (u32x4){0, 0, 0, 0};
            f[g].q[1] = (u32x4){0, 0, 0, 0};
          }
        }
      } else {                                // K tail: only k=192..199 valid
#pragma unroll
        for (int g = 0; g < 4; ++g) {
          f[g].q[0] = (u32x4){0, 0, 0, 0};
          f[g].q[1] = (u32x4){0, 0, 0, 0};
          if (jok && kb8 == 0)
            f[g].q[0] = *(const u32x4*)(wbase + (size_t)g * HH * HH + 192);
        }
      }
#pragma unroll
      for (int g = 0; g < 4; ++g) acc[g] = wmma_bf16(a, f[g].v, acc[g]);
    }

    // ---- prefetch pre for t+1 into registers (independent of barriers) ----
    {
      int tn = (t + 1 < TT) ? (t + 1) : t;
      __builtin_prefetch(&pre[((size_t)tn * BB + b0) * G4], 0, 1);
#pragma unroll
      for (int g = 0; g < 4; ++g)
#pragma unroll
        for (int r = 0; r < 8; ++r)
          prn[g][r] = jok ? pre[((size_t)tn * BB + b0 + r + 8 * hi) * G4 +
                                g * HH + j]
                          : 0.0f;
    }

    __syncthreads();  // all waves done reading hlds[t]

    // ---- gate nonlinearity + state update; all 4 gates are lane-local ----
#pragma unroll
    for (int r = 0; r < 8; ++r) {
      float gi = sigmoid_f(acc[0][r] + pr[0][r]);
      float gf = sigmoid_f(acc[1][r] + pr[1][r]);
      float gg = tanh_f   (acc[2][r] + pr[2][r]);
      float go = sigmoid_f(acc[3][r] + pr[3][r]);
      float cn = gf * c[r] + gi * gg;
      c[r] = cn;
      float h = go * tanh_f(cn);
      if (jok) {
        hlds[(r + 8 * hi) * 232 + j] = (__bf16)h;   // pad cols stay zero
        op[r] += h * Wout[t * HH + j];              // fused output projection
      }
    }
#pragma unroll
    for (int g = 0; g < 4; ++g)
#pragma unroll
      for (int r = 0; r < 8; ++r) pr[g][r] = prn[g][r];

    __syncthreads();  // hlds[t+1] visible to every wave
  }

  // ---- final reduction of output partials over j (lanes + waves) ----
  if (jok) {
#pragma unroll
    for (int r = 0; r < 8; ++r) atomicAdd(&oacc[r + 8 * hi], op[r]);
  }
  __syncthreads();
  if (tid < 16) out[b0 + tid] = oacc[tid] + bout[0];
}

// ---------------------------------------------------------------------- launch
extern "C" void kernel_launch(void* const* d_in, const int* in_sizes, int n_in,
                              void* d_out, int out_size, void* d_ws,
                              size_t ws_size, hipStream_t stream) {
  (void)in_sizes; (void)n_in; (void)out_size; (void)ws_size;
  const float* x     = (const float*)d_in[0];  // [B,T,D]
  const float* W_ih  = (const float*)d_in[1];  // [800,256]
  const float* W_hh  = (const float*)d_in[2];  // [800,200]
  const float* b_ih  = (const float*)d_in[3];  // [800]
  const float* b_hh  = (const float*)d_in[4];  // [800]
  const float* W_out = (const float*)d_in[5];  // [1, T*H]
  const float* b_out = (const float*)d_in[6];  // [1]
  float* out = (float*)d_out;                  // [B]

  char* ws = (char*)d_ws;
  size_t off = 0;
  __bf16* xb      = (__bf16*)(ws + off); off += (size_t)BB * TT * DD * 2;  // 128 MiB
  __bf16* Wihb    = (__bf16*)(ws + off); off += (size_t)G4 * DD * 2;       // 400 KiB
  __bf16* Whhb    = (__bf16*)(ws + off); off += (size_t)G4 * HH * 2;       // 312.5 KiB
  float*  biasSum = (float*)(ws + off);  off += 4096;                      // 800 f32 + pad
  float*  pre     = (float*)(ws + off);  // [T][B][800] f32 = 800 MiB

  // 0) precision conversion + bias fold
  {
    int n = BB * TT * DD;
    f32_to_bf16_kernel<<<dim3(8192), dim3(256), 0, stream>>>(x, xb, n);
    f32_to_bf16_kernel<<<dim3(512), dim3(256), 0, stream>>>(W_ih, Wihb, G4 * DD);
    f32_to_bf16_kernel<<<dim3(512), dim3(256), 0, stream>>>(W_hh, Whhb, G4 * HH);
    bias_sum_kernel<<<dim3(4), dim3(256), 0, stream>>>(b_ih, b_hh, biasSum, G4);
  }
  // 1) pre = x @ W_ih^T + (b_ih + b_hh), stored [T][B][800]
  pre_gemm_kernel<<<dim3((BB * TT) / 16), dim3(320), 0, stream>>>(
      xb, Wihb, biasSum, pre);
  // 2) recurrence + fused output projection (W_hh resident in LDS via TDM)
  lstm_kernel<<<dim3(BB / 16), dim3(416), 0, stream>>>(
      Whhb, pre, W_out, b_out, out);
}
